// LASAGE_39822936769198
// MI455X (gfx1250) — compile-verified, matrix operand order checked by reading, not run on hardware
//
#include <hip/hip_runtime.h>

// CDNA5 / gfx1250: wave32, WMMA 16x16x32 bf16, B staged in LDS, templated GEMM.

typedef __attribute__((ext_vector_type(16))) __bf16 v16bf;
typedef __attribute__((ext_vector_type(8)))  float  v8f;

__device__ __forceinline__ __bf16 f2bf(float f) {
  return (__bf16)f;   // native v_cvt_pk_bf16_f32 on gfx1250, RNE
}

// ---------------- degree / scale ----------------
__global__ void k_deg(const long long* __restrict__ dst, float* __restrict__ deg, int E) {
  int i = blockIdx.x * blockDim.x + threadIdx.x;
  if (i < E) atomicAdd(&deg[(int)dst[i]], 1.0f);
}

__global__ void k_scale(const float* __restrict__ deg, float* __restrict__ scl, int n) {
  int i = blockIdx.x * blockDim.x + threadIdx.x;
  if (i < n) scl[i] = 1.0f / fmaxf(deg[i], 1.0f);
}

// ---------------- edge scatter-add (wave per edge) ----------------
__global__ void k_aggregate(const float* __restrict__ x,
                            const long long* __restrict__ src,
                            const long long* __restrict__ dst,
                            float* __restrict__ agg, int E, int d) {
  int w    = (blockIdx.x * blockDim.x + threadIdx.x) >> 5;   // wave32
  int lane = threadIdx.x & 31;
  if (w >= E) return;
  int s = (int)src[w];
  int t = (int)dst[w];
  const float* xs = x + (size_t)s * d;
  float*       at = agg + (size_t)t * d;
  for (int j = lane; j < d; j += 32) atomicAdd(&at[j], xs[j]);
}

// ---------------- pack [Wl;Wr] (2K x D fp32) into bf16 WMMA B-fragment order ----
// Fragment layout (16-bit B 32x16): lane -> col n = lane&15, half = lane>>4,
// element j -> k = kb*32 + (j>>3)*16 + half*8 + (j&7).
__global__ void k_pack_w(const float* __restrict__ Wl, const float* __restrict__ Wr,
                         __bf16* __restrict__ Bp, int K, int D) {
  int Kb = (2 * K) >> 5;
  int Nt = D >> 4;
  int idx  = blockIdx.x * blockDim.x + threadIdx.x;
  int lane = idx & 31;
  int tile = idx >> 5;
  if (tile >= Kb * Nt) return;
  int nt = tile % Nt, kb = tile / Nt;
  int n    = (nt << 4) + (lane & 15);
  int half = lane >> 4;
  v16bf frag;
#pragma unroll
  for (int j = 0; j < 16; ++j) {
    int k = (kb << 5) + ((j >> 3) << 4) + (half << 3) + (j & 7);
    float w = (k < K) ? Wl[(size_t)k * D + n] : Wr[(size_t)(k - K) * D + n];
    frag[j] = f2bf(w);
  }
  *(v16bf*)(Bp + ((size_t)tile * 32 + lane) * 16) = frag;
}

// ---------------- one K-step: 32-deep A fragment, 4 WMMAs against LDS B ----
__device__ __forceinline__ void mm_step(const float* __restrict__ sp, float mul,
                                        int kb, v8f acc[4],
                                        const __bf16* __restrict__ lbs, int lane) {
  float4 c0 = *(const float4*)(sp);
  float4 c1 = *(const float4*)(sp + 4);
  float4 c2 = *(const float4*)(sp + 16);
  float4 c3 = *(const float4*)(sp + 20);
  v16bf af;
  af[0]  = f2bf(c0.x * mul); af[1]  = f2bf(c0.y * mul);
  af[2]  = f2bf(c0.z * mul); af[3]  = f2bf(c0.w * mul);
  af[4]  = f2bf(c1.x * mul); af[5]  = f2bf(c1.y * mul);
  af[6]  = f2bf(c1.z * mul); af[7]  = f2bf(c1.w * mul);
  af[8]  = f2bf(c2.x * mul); af[9]  = f2bf(c2.y * mul);
  af[10] = f2bf(c2.z * mul); af[11] = f2bf(c2.w * mul);
  af[12] = f2bf(c3.x * mul); af[13] = f2bf(c3.y * mul);
  af[14] = f2bf(c3.z * mul); af[15] = f2bf(c3.w * mul);
#pragma unroll
  for (int t = 0; t < 4; ++t) {
    v16bf bf = *(const v16bf*)(lbs + ((size_t)(kb * 4 + t) * 32 + lane) * 16);
    acc[t] = __builtin_amdgcn_wmma_f32_16x16x32_bf16(
        /*neg_a=*/false, af, /*neg_b=*/false, bf,
        /*c_mod=*/(short)0, acc[t], /*reuse_a=*/false, /*reuse_b=*/false);
  }
}

// ---------------- fused SAGE GEMM (templated on K, D) ----------------
// out = (agg/deg)@Wl + x@Wr + b [, relu]
// Block = 8 waves sharing one 64-column group (4 n-tiles), covering 8 row-tiles.
// B fragments for all k-steps of the column group staged once in LDS.
template <int K, int D>
__global__ void k_sage_gemm(const float* __restrict__ agg,
                            const float* __restrict__ x,
                            const float* __restrict__ scl,
                            const __bf16* __restrict__ Bp,
                            const float* __restrict__ bias,
                            float* __restrict__ out,
                            int M, int ldo, int col_off, int relu) {
  constexpr int Nt  = D / 16;
  constexpr int Kb  = (2 * K) / 32;       // total k-steps
  constexpr int KbH = K / 32;             // k-steps per half (agg / x)
  constexpr int ntGroups = Nt / 4;
  __shared__ __bf16 lbs[Kb * 4 * 512];    // [kb][t][lane][16]

  const int ntg = (blockIdx.x % ntGroups) << 2;
  const int mtb = blockIdx.x / ntGroups;
  const int wid  = threadIdx.x >> 5;
  const int lane = threadIdx.x & 31;

  // cooperative copy of this block's B slice: Kb*4 tiles x 1KB each
#pragma unroll
  for (int c = threadIdx.x; c < Kb * 256; c += 256) {
    int blk = c >> 6, within = c & 63;    // 64 x float4 per 1KB tile
    int kb = blk >> 2, t = blk & 3;
    const float4* g = (const float4*)(Bp + (size_t)(kb * Nt + ntg + t) * 512) + within;
    ((float4*)(lbs + (size_t)blk * 512))[within] = *g;
  }
  __syncthreads();

  const int mt = mtb * 8 + wid;
  if (mt * 16 >= M) return;               // wave-uniform: EXEC stays all-1s

  const int   row  = mt * 16 + (lane & 15);   // A: lane -> row M
  const int   half = lane >> 4;               // A: K-octet select
  const float rs   = scl[row];
  const float* ap = agg + (size_t)row * K + half * 8;
  const float* xp = x   + (size_t)row * K + half * 8;

  v8f zero = {};
  v8f acc[4];
#pragma unroll
  for (int t = 0; t < 4; ++t) acc[t] = zero;

  // agg half: scaled by 1/deg
#pragma unroll 4
  for (int kb = 0; kb < KbH; ++kb)
    mm_step(ap + kb * 32, rs, kb, acc, lbs, lane);
  // x half: multiplier folds to 1.0
#pragma unroll 4
  for (int kb = 0; kb < KbH; ++kb)
    mm_step(xp + kb * 32, 1.0f, KbH + kb, acc, lbs, lane);

#pragma unroll
  for (int t = 0; t < 4; ++t) {
    int   ncol = (ntg + t) * 16 + (lane & 15);
    float bv   = bias[ncol];
#pragma unroll
    for (int r = 0; r < 8; ++r) {
      int   m = mt * 16 + half * 8 + r;   // C/D layout: VGPR r -> row half*8+r
      float v = acc[t][r] + bv;
      if (relu) v = fmaxf(v, 0.0f);
      out[(size_t)m * ldo + col_off + ncol] = v;
    }
  }
}

// ---------------- host orchestration ----------------
extern "C" void kernel_launch(void* const* d_in, const int* in_sizes, int n_in,
                              void* d_out, int out_size, void* d_ws, size_t ws_size,
                              hipStream_t stream) {
  const float*     x0  = (const float*)d_in[0];
  const float*     x1  = (const float*)d_in[1];
  const long long* ei  = (const long long*)d_in[2];
  const float*     Wl0 = (const float*)d_in[3];
  const float*     Wr0 = (const float*)d_in[4];
  const float*     b0  = (const float*)d_in[5];
  const float*     Wl1 = (const float*)d_in[6];
  const float*     Wr1 = (const float*)d_in[7];
  const float*     b1  = (const float*)d_in[8];
  const float*     Wlm = (const float*)d_in[9];
  const float*     Wrm = (const float*)d_in[10];
  const float*     bm  = (const float*)d_in[11];
  const float*     Wlo = (const float*)d_in[12];
  const float*     Wro = (const float*)d_in[13];
  const float*     bo  = (const float*)d_in[14];

  const int D_IN = 128, D_MID = 256;
  const int Nn = in_sizes[0] / D_IN;    // 50000
  const int E  = in_sizes[2] / 2;       // 800000
  const long long* srcI = ei;
  const long long* dstI = ei + E;

  // workspace partition (~154 MB)
  char*  ws  = (char*)d_ws;
  size_t off = 0;
  auto alloc = [&](size_t bytes) {
    char* p = ws + off;
    off += (bytes + 255) & ~(size_t)255;
    return p;
  };
  float*  deg  = (float*)alloc((size_t)Nn * 4);
  float*  scl  = (float*)alloc((size_t)Nn * 4);
  float*  aggb = (float*)alloc((size_t)Nn * D_MID * 4);   // reused per layer
  float*  hbuf = (float*)alloc((size_t)Nn * D_MID * 4);   // concat(h0, h1)
  float*  hmid = (float*)alloc((size_t)Nn * D_MID * 4);
  __bf16* Bp   = (__bf16*)alloc((size_t)(2 * D_MID) * D_MID * 2);
  (void)ws_size; (void)n_in; (void)out_size;

  float* outp = (float*)d_out;
  const int TB = 256;
  const int Mt = (Nn + 15) / 16;

  hipMemsetAsync(deg, 0, (size_t)Nn * 4, stream);
  k_deg  <<<(E  + TB - 1) / TB, TB, 0, stream>>>(dstI, deg, E);
  k_scale<<<(Nn + TB - 1) / TB, TB, 0, stream>>>(deg, scl, Nn);

  // aggregation + weight-pack common to each layer
  auto prep = [&](const float* xin, int K, const float* Wl, const float* Wr, int D) {
    hipMemsetAsync(aggb, 0, (size_t)Nn * K * 4, stream);
    k_aggregate<<<(E * 32 + TB - 1) / TB, TB, 0, stream>>>(xin, srcI, dstI, aggb, E, K);
    int packThreads = ((2 * K) / 32) * (D / 16) * 32;
    k_pack_w<<<(packThreads + TB - 1) / TB, TB, 0, stream>>>(Wl, Wr, Bp, K, D);
  };
  auto blocks = [&](int D) { return (D / 64) * ((Mt + 7) / 8); };

  // layer 0, branch 0: 128 -> 128, relu -> hbuf[:, 0:128]
  prep(x0, 128, Wl0, Wr0, 128);
  k_sage_gemm<128, 128><<<blocks(128), TB, 0, stream>>>(
      aggb, x0, scl, Bp, b0, hbuf, Nn, 256, 0, 1);
  // layer 0, branch 1: 128 -> 128, relu -> hbuf[:, 128:256]  (concat is free)
  prep(x1, 128, Wl1, Wr1, 128);
  k_sage_gemm<128, 128><<<blocks(128), TB, 0, stream>>>(
      aggb, x1, scl, Bp, b1, hbuf, Nn, 256, 128, 1);
  // middle conv: 256 -> 256, relu
  prep(hbuf, 256, Wlm, Wrm, 256);
  k_sage_gemm<256, 256><<<blocks(256), TB, 0, stream>>>(
      aggb, hbuf, scl, Bp, bm, hmid, Nn, 256, 0, 1);
  // final conv: 256 -> 64, no activation, straight to d_out
  prep(hmid, 256, Wlo, Wro, 64);
  k_sage_gemm<256, 64><<<blocks(64), TB, 0, stream>>>(
      aggb, hmid, scl, Bp, bo, outp, Nn, 64, 0, 0);
}
